// DiffPoolBatchedGraphLayer_29016799052037
// MI455X (gfx1250) — matrix operand order repaired and checked by reading, not compiled, not run on hardware
//
#include <hip/hip_runtime.h>
#include <math.h>

// ---------------- problem constants (from reference setup_inputs) ----------
#define NGRAPH 64      // B
#define NNODE  1024    // n  (shift 10)
#define FDIM   128     // F
#define KDIM   128     // K
#define NEDGE  16384   // E  (shift 14)
#define NTOT   (NGRAPH * NNODE)   // 65536 nodes
#define BE     (NGRAPH * NEDGE)   // 1048576 edges
#define ADJ_ELEMS ((long)(NGRAPH * KDIM) * (long)(NGRAPH * KDIM)) // 67,108,864

typedef __attribute__((ext_vector_type(16))) __bf16 v16bf;
typedef __attribute__((ext_vector_type(8)))  __bf16 v8bf;
typedef __attribute__((ext_vector_type(8)))  float  v8f;

union BF16x16 { v16bf v; struct { v8bf lo, hi; } s; };

__device__ __forceinline__ float waveReduceSum(float v) {
  #pragma unroll
  for (int m = 16; m >= 1; m >>= 1) v += __shfl_xor(v, m, 32);
  return v;
}
__device__ __forceinline__ float waveReduceMax(float v) {
  #pragma unroll
  for (int m = 16; m >= 1; m >>= 1) v = fmaxf(v, __shfl_xor(v, m, 32));
  return v;
}

// ---------------------------------------------------------------------------
// zero a float4-aligned region
__global__ void zero4_kernel(float4* __restrict__ p, long n4) {
  long i = (long)blockIdx.x * blockDim.x + threadIdx.x;
  if (i < n4) p[i] = make_float4(0.f, 0.f, 0.f, 0.f);
}

// ---------------------------------------------------------------------------
// Pack W_feat / W_pool [256x128] f32 -> bf16 in WMMA B-fragment order:
// pack[((ct*8+kt)*32+lane)*16 + j] = W[kt*32 + (lane<16?0:16) + j][ct*16 + (lane&15)]
__global__ void pack_w_kernel(const float* __restrict__ Wf, const float* __restrict__ Wp,
                              __bf16* __restrict__ Pf, __bf16* __restrict__ Pp) {
  int idx = blockIdx.x * blockDim.x + threadIdx.x;     // 0 .. 65535
  int which = idx >> 15;
  int r = idx & 32767;
  int ct   = (r >> 12) & 7;
  int kt   = (r >> 9)  & 7;
  int lane = (r >> 4)  & 31;
  int j    =  r        & 15;
  int k  = kt * 32 + ((lane < 16) ? 0 : 16) + j;
  int nn = ct * 16 + (lane & 15);
  const float* W = which ? Wp : Wf;
  __bf16*      P = which ? Pp : Pf;
  P[r] = (__bf16)W[k * FDIM + nn];
}

// ---------------------------------------------------------------------------
// degree: deg[dst] += 1 per edge
__global__ void deg_kernel(const int* __restrict__ dst_local, float* __restrict__ deg) {
  int e = blockIdx.x * blockDim.x + threadIdx.x;
  if (e >= BE) return;
  int g = e >> 14;
  int d = dst_local[e] + (g << 10);
  atomicAdd(&deg[d], 1.0f);
}

// c_sum[dst] += h[src] (128 floats/edge, 4 per thread)
__global__ void cscatter_kernel(const float* __restrict__ h,
                                const int* __restrict__ src_local,
                                const int* __restrict__ dst_local,
                                float* __restrict__ c_sum) {
  long gid = (long)blockIdx.x * blockDim.x + threadIdx.x;
  long e = gid >> 5;
  int  q = (int)(gid & 31);
  if (e >= BE) return;
  int g = (int)(e >> 14);
  int s = src_local[e] + (g << 10);
  int d = dst_local[e] + (g << 10);
  float4 v = *(const float4*)(h + (long)s * FDIM + q * 4);
  float* out = c_sum + (long)d * FDIM + q * 4;
  atomicAdd(out + 0, v.x);
  atomicAdd(out + 1, v.y);
  atomicAdd(out + 2, v.z);
  atomicAdd(out + 3, v.w);
}

// AS[src] += assign[dst] (gather bf16 row-major assign)
__global__ void asscatter_kernel(const __bf16* __restrict__ assign_rm,
                                 const int* __restrict__ src_local,
                                 const int* __restrict__ dst_local,
                                 float* __restrict__ AS) {
  long gid = (long)blockIdx.x * blockDim.x + threadIdx.x;
  long e = gid >> 5;
  int  q = (int)(gid & 31);
  if (e >= BE) return;
  int g = (int)(e >> 14);
  int s = src_local[e] + (g << 10);
  int d = dst_local[e] + (g << 10);
  const __bf16* a = assign_rm + (long)d * KDIM + q * 4;
  float* out = AS + (long)s * KDIM + q * 4;
  atomicAdd(out + 0, (float)a[0]);
  atomicAdd(out + 1, (float)a[1]);
  atomicAdd(out + 2, (float)a[2]);
  atomicAdd(out + 3, (float)a[3]);
}

// AS f32 [N][128] -> AST bf16 [g][128][1024]
__global__ void astconv_kernel(const float* __restrict__ AS, __bf16* __restrict__ AST) {
  long i = (long)blockIdx.x * blockDim.x + threadIdx.x;
  if (i >= (long)NTOT * KDIM) return;
  int node = (int)(i >> 7);
  int col  = (int)(i & 127);
  int g  = node >> 10;
  int nl = node & 1023;
  AST[((long)(g * KDIM + col) << 10) + nl] = (__bf16)AS[i];
}

// ---------------------------------------------------------------------------
// Fused SAGE: z = [h | c] @ W + b for both W_feat and W_pool via WMMA,
// then l2norm+relu (+softmax for pool). 16 nodes per block, 8 waves.
__global__ __launch_bounds__(256) void sage_kernel(
    const float* __restrict__ h, const float* __restrict__ c_sum,
    const float* __restrict__ deg,
    const __bf16* __restrict__ Wf, const __bf16* __restrict__ Wp,
    const float* __restrict__ bfeat, const float* __restrict__ bpool,
    __bf16* __restrict__ featT, __bf16* __restrict__ assignT,
    __bf16* __restrict__ assign_rm)
{
  __shared__ __bf16 Ash[16][256];   // [node row][k of concat(h,c)] bf16
  __shared__ float  zf[16][128];
  __shared__ float  zp[16][128];

  const int tid = threadIdx.x;
  const int nodeBase = blockIdx.x << 4;

  // ---- stage concat(h, c) as bf16 into LDS ----
  {
    int row = tid >> 4;
    int colBase = (tid & 15) << 4;           // 0,16,...,240
    int node = nodeBase + row;
    float inv = 1.0f;
    const float* src;
    if (colBase < FDIM) {
      src = h + (long)node * FDIM + colBase;
    } else {
      inv = 1.0f / fmaxf(deg[node], 1.0f);
      src = c_sum + (long)node * FDIM + (colBase - FDIM);
    }
    #pragma unroll
    for (int j = 0; j < 4; ++j) {
      float4 v = ((const float4*)src)[j];
      Ash[row][colBase + j * 4 + 0] = (__bf16)(v.x * inv);
      Ash[row][colBase + j * 4 + 1] = (__bf16)(v.y * inv);
      Ash[row][colBase + j * 4 + 2] = (__bf16)(v.z * inv);
      Ash[row][colBase + j * 4 + 3] = (__bf16)(v.w * inv);
    }
  }
  __syncthreads();

  const int wave = tid >> 5;
  const int lane = tid & 31;
  const int arow  = lane & 15;
  const int akoff = (lane < 16) ? 0 : 16;

  v8f accF = {0.f,0.f,0.f,0.f,0.f,0.f,0.f,0.f};
  v8f accP = {0.f,0.f,0.f,0.f,0.f,0.f,0.f,0.f};

  #pragma unroll
  for (int kt = 0; kt < 8; ++kt) {
    BF16x16 a, wfv, wpv;
    a.s.lo = *(const v8bf*)&Ash[arow][kt * 32 + akoff];
    a.s.hi = *(const v8bf*)&Ash[arow][kt * 32 + akoff + 8];
    const __bf16* pf = Wf + (((wave * 8 + kt) * 32 + lane) << 4);
    wfv.s.lo = *(const v8bf*)pf;
    wfv.s.hi = *(const v8bf*)(pf + 8);
    const __bf16* pp = Wp + (((wave * 8 + kt) * 32 + lane) << 4);
    wpv.s.lo = *(const v8bf*)pp;
    wpv.s.hi = *(const v8bf*)(pp + 8);
    accF = __builtin_amdgcn_wmma_f32_16x16x32_bf16(false, a.v, false, wfv.v,
                                                   (short)0, accF, false, false);
    accP = __builtin_amdgcn_wmma_f32_16x16x32_bf16(false, a.v, false, wpv.v,
                                                   (short)0, accP, false, false);
  }

  // ---- spill z tiles (+bias) to LDS ----
  {
    int cN = (wave << 4) + (lane & 15);
    int rOff = (lane < 16) ? 0 : 8;
    float bF = bfeat[cN], bP = bpool[cN];
    #pragma unroll
    for (int i = 0; i < 8; ++i) {
      zf[rOff + i][cN] = accF[i] + bF;
      zp[rOff + i][cN] = accP[i] + bP;
    }
  }
  __syncthreads();

  // ---- per-row l2norm + relu (+softmax); one wave handles 2 rows ----
  for (int rr = 0; rr < 2; ++rr) {
    int row  = (wave << 1) + rr;
    int node = nodeBase + row;
    int g  = node >> 10;
    int nl = node & 1023;
    int col = lane << 2;

    // feat = relu(l2norm(zf))
    {
      float4 v = *(float4*)&zf[row][col];
      float ss = waveReduceSum(v.x*v.x + v.y*v.y + v.z*v.z + v.w*v.w);
      float sc = 1.0f / fmaxf(sqrtf(ss), 1e-12f);
      float f0 = fmaxf(v.x * sc, 0.f), f1 = fmaxf(v.y * sc, 0.f);
      float f2 = fmaxf(v.z * sc, 0.f), f3 = fmaxf(v.w * sc, 0.f);
      featT[((long)(g * FDIM + col + 0) << 10) + nl] = (__bf16)f0;
      featT[((long)(g * FDIM + col + 1) << 10) + nl] = (__bf16)f1;
      featT[((long)(g * FDIM + col + 2) << 10) + nl] = (__bf16)f2;
      featT[((long)(g * FDIM + col + 3) << 10) + nl] = (__bf16)f3;
    }
    // assign = softmax(relu(l2norm(zp)))
    {
      float4 v = *(float4*)&zp[row][col];
      float ss = waveReduceSum(v.x*v.x + v.y*v.y + v.z*v.z + v.w*v.w);
      float sc = 1.0f / fmaxf(sqrtf(ss), 1e-12f);
      float p0 = fmaxf(v.x * sc, 0.f), p1 = fmaxf(v.y * sc, 0.f);
      float p2 = fmaxf(v.z * sc, 0.f), p3 = fmaxf(v.w * sc, 0.f);
      float mx = waveReduceMax(fmaxf(fmaxf(p0, p1), fmaxf(p2, p3)));
      float e0 = __expf(p0 - mx), e1 = __expf(p1 - mx);
      float e2 = __expf(p2 - mx), e3 = __expf(p3 - mx);
      float es = waveReduceSum(e0 + e1 + e2 + e3);
      float inv = 1.0f / es;
      float a0 = e0 * inv, a1 = e1 * inv, a2 = e2 * inv, a3 = e3 * inv;
      assignT[((long)(g * KDIM + col + 0) << 10) + nl] = (__bf16)a0;
      assignT[((long)(g * KDIM + col + 1) << 10) + nl] = (__bf16)a1;
      assignT[((long)(g * KDIM + col + 2) << 10) + nl] = (__bf16)a2;
      assignT[((long)(g * KDIM + col + 3) << 10) + nl] = (__bf16)a3;
      __bf16* rm = assign_rm + (long)node * KDIM + col;
      rm[0] = (__bf16)a0; rm[1] = (__bf16)a1; rm[2] = (__bf16)a2; rm[3] = (__bf16)a3;
    }
  }
}

// ---------------------------------------------------------------------------
// Per-graph C[128x128] = A_g[128x1024] * B_g[1024x128] with A,B stored as
// [g][row/col 128][node 1024] bf16. One wave per 16x16 tile, K=1024.
__global__ __launch_bounds__(256) void stmm_kernel(
    const __bf16* __restrict__ A, const __bf16* __restrict__ Bm,
    float* __restrict__ out, long outStride, int colPerGraph)
{
  int gtile = blockIdx.x * 8 + (threadIdx.x >> 5);   // 4096 tiles total
  int g = gtile >> 6;
  int t = gtile & 63;
  int m0 = (t >> 3) << 4;
  int n0 = (t & 7) << 4;
  int lane = threadIdx.x & 31;

  const __bf16* Ap = A  + (((long)g * KDIM + m0 + (lane & 15)) << 10) + ((lane < 16) ? 0 : 16);
  const __bf16* Bp = Bm + (((long)g * FDIM + n0 + (lane & 15)) << 10) + ((lane < 16) ? 0 : 16);

  v8f acc = {0.f,0.f,0.f,0.f,0.f,0.f,0.f,0.f};
  #pragma unroll 4
  for (int kt = 0; kt < 32; ++kt) {
    BF16x16 a, b;
    a.s.lo = *(const v8bf*)(Ap + kt * 32);
    a.s.hi = *(const v8bf*)(Ap + kt * 32 + 8);
    b.s.lo = *(const v8bf*)(Bp + kt * 32);
    b.s.hi = *(const v8bf*)(Bp + kt * 32 + 8);
    acc = __builtin_amdgcn_wmma_f32_16x16x32_bf16(false, a.v, false, b.v,
                                                  (short)0, acc, false, false);
  }
  long rowBase = (long)g * KDIM + m0 + ((lane < 16) ? 0 : 8);
  long colBase = (colPerGraph ? (long)g * KDIM : 0) + n0 + (lane & 15);
  #pragma unroll
  for (int i = 0; i < 8; ++i)
    out[(rowBase + i) * outStride + colBase] = acc[i];
}

// ---------------------------------------------------------------------------
extern "C" void kernel_launch(void* const* d_in, const int* in_sizes, int n_in,
                              void* d_out, int out_size, void* d_ws, size_t ws_size,
                              hipStream_t stream) {
  const float* h         = (const float*)d_in[0];
  const int*   src_local = (const int*)  d_in[1];
  const int*   dst_local = (const int*)  d_in[2];
  const float* W_feat    = (const float*)d_in[3];
  const float* b_feat    = (const float*)d_in[4];
  const float* W_pool    = (const float*)d_in[5];
  const float* b_pool    = (const float*)d_in[6];
  float* out = (float*)d_out;

  // ---- workspace layout ----
  char* ws = (char*)d_ws;
  size_t off = 0;
  float* deg = (float*)(ws + off);            off += (size_t)NTOT * 4;                 // 256 KB
  float* c_sum = (float*)(ws + off);          // aliased with AS (re-zeroed)
  float* AS    = c_sum;                       off += (size_t)NTOT * FDIM * 4;          // 32 MB
  __bf16* featT    = (__bf16*)(ws + off);     off += (size_t)NGRAPH * FDIM * NNODE * 2; // 16 MB
  __bf16* assignT  = (__bf16*)(ws + off);     off += (size_t)NGRAPH * KDIM * NNODE * 2; // 16 MB
  __bf16* assign_rm= (__bf16*)(ws + off);     off += (size_t)NTOT * KDIM * 2;           // 16 MB
  __bf16* AST      = (__bf16*)(ws + off);     off += (size_t)NGRAPH * KDIM * NNODE * 2; // 16 MB
  __bf16* Wf_pack  = (__bf16*)(ws + off);     off += (size_t)256 * FDIM * 2;            // 64 KB
  __bf16* Wp_pack  = (__bf16*)(ws + off);     off += (size_t)256 * KDIM * 2;            // 64 KB
  (void)off; (void)ws_size; (void)in_sizes; (void)n_in; (void)out_size;

  const int TPB = 256;

  // 1) zero deg + c_sum (contiguous region)
  {
    long n4 = ((long)NTOT * 4 + (long)NTOT * FDIM * 4) / 16;
    zero4_kernel<<<(unsigned)((n4 + TPB - 1) / TPB), TPB, 0, stream>>>((float4*)deg, n4);
  }
  // 2) pack weights to bf16 fragments
  pack_w_kernel<<<65536 / TPB, TPB, 0, stream>>>(W_feat, W_pool, Wf_pack, Wp_pack);
  // 3) degrees
  deg_kernel<<<BE / TPB, TPB, 0, stream>>>(dst_local, deg);
  // 4) c_sum scatter
  cscatter_kernel<<<(unsigned)(((long)BE * 32) / TPB), TPB, 0, stream>>>(h, src_local, dst_local, c_sum);
  // 5) fused SAGE (WMMA GEMM + l2norm/relu/softmax)
  sage_kernel<<<NTOT / 16, TPB, 0, stream>>>(h, c_sum, deg, Wf_pack, Wp_pack,
                                             b_feat, b_pool, featT, assignT, assign_rm);
  // 6) re-zero aliased AS buffer
  {
    long n4 = ((long)NTOT * KDIM * 4) / 16;
    zero4_kernel<<<(unsigned)((n4 + TPB - 1) / TPB), TPB, 0, stream>>>((float4*)AS, n4);
  }
  // 7) AS scatter: AS[src] += assign[dst]
  asscatter_kernel<<<(unsigned)(((long)BE * 32) / TPB), TPB, 0, stream>>>(assign_rm, src_local, dst_local, AS);
  // 8) AS -> AST (bf16, graph-transposed)
  astconv_kernel<<<(unsigned)(((long)NTOT * KDIM + TPB - 1) / TPB), TPB, 0, stream>>>(AS, AST);
  // 9) zero adjacency output (256 MB; diagonal blocks overwritten below)
  {
    long n4 = ADJ_ELEMS / 4;
    zero4_kernel<<<(unsigned)((n4 + TPB - 1) / TPB), TPB, 0, stream>>>((float4*)out, n4);
  }
  // 10) h_new = S^T feat  -> out[ADJ_ELEMS ...], row stride 128
  stmm_kernel<<<(NGRAPH * 64) / 8, TPB, 0, stream>>>(assignT, featT,
                                                     out + ADJ_ELEMS, (long)FDIM, 0);
  // 11) diagonal blocks of adj_new = S^T AS, row stride 8192, col offset g*128
  stmm_kernel<<<(NGRAPH * 64) / 8, TPB, 0, stream>>>(assignT, AST,
                                                     out, (long)(NGRAPH * KDIM), 1);
}
